// EfficientFormerSelfAttention_16655883174062
// MI455X (gfx1250) — matrix operand order repaired and checked by reading, not compiled, not run on hardware
//
#include <hip/hip_runtime.h>
#include <cstdint>
#include <cstddef>

// ---------------------------------------------------------------------------
// EfficientFormer self-attention, fused, bf16 WMMA (gfx1250 / CDNA5, wave32)
// 4 m-tiles x 2 n-tiles per wave: 4 A-fragment LDS loads feed 8 WMMAs/k-step.
// ---------------------------------------------------------------------------

typedef __attribute__((ext_vector_type(16))) __bf16 v16bf;
typedef __attribute__((ext_vector_type(8)))  float  v8f;

using u16 = unsigned short;
using u32 = unsigned int;

// ---- problem constants ----
#define NB    2048      // batch
#define SQ    49        // sequence (7x7)
#define SP    64        // padded sequence
#define DIN   448       // model dim
#define HID   1536      // qkv hidden = 8*(32+32+128)
#define NH    8         // heads
#define QKVD  192       // per-head qkv width
#define ED    128       // expanded (v/ctx) dim per head
#define PROJK 1024      // NH*ED

// ---- LDS strides (elements). All fragment offsets stay 16B-aligned. ----
#define XS_ST   456     // 448 + 8 halfs
#define QKV_ST  1544    // 1536 + 8 halfs
#define P_ST    72      // 64 + 8
#define SMEM_BYTES (SP*XS_ST*2 + SP*QKV_ST*2 + SP*P_ST*4 + SP*P_ST*2 + ED*P_ST*2) // 302080

// ---- workspace layout (bytes) ----
#define WS_QKVW  0u                         // 1536*448 bf16 = 1,376,256 B
#define WS_PROJW 1376256u                   // 448*1024 bf16 =   917,504 B
#define WS_BIAS  2293760u                   // 8*49*49  f32  =    76,832 B

__device__ __forceinline__ u16 f2bf(float x) {
    u32 u = __builtin_bit_cast(u32, x);
    u32 r = u + 0x7FFFu + ((u >> 16) & 1u);   // round-to-nearest-even
    return (u16)(r >> 16);
}

// 16 bf16 values = two 16-byte chunks (matches the per-lane A/B chunking)
__device__ __forceinline__ v16bf ldfrag(const u16* p0, const u16* p1) {
    union { uint4 q[2]; v16bf v; } u;
    u.q[0] = *(const uint4*)p0;
    u.q[1] = *(const uint4*)p1;
    return u.v;
}

__device__ __forceinline__ v8f wmma_bf16(v16bf a, v16bf b, v8f c) {
    // v_wmma_f32_16x16x32_bf16  D = A(16x32) * B(32x16) + C
    return __builtin_amdgcn_wmma_f32_16x16x32_bf16(
        /*neg_a=*/false, a, /*neg_b=*/false, b,
        /*c_mod=*/(short)0, c, /*reuse_a=*/false, /*reuse_b=*/false);
}

__device__ __forceinline__ v8f vzero() {
    v8f z = {0.f, 0.f, 0.f, 0.f, 0.f, 0.f, 0.f, 0.f};
    return z;
}

// ---------------------------------------------------------------------------
// Prep: fp32 weights -> bf16 (L2-resident, reused by all 2048 blocks)
// ---------------------------------------------------------------------------
__global__ void prep_weights(const float* __restrict__ qw,
                             const float* __restrict__ pw,
                             u16* __restrict__ qwb,
                             u16* __restrict__ pwb) {
    int i = blockIdx.x * 256 + threadIdx.x;
    if (i < HID * DIN)   qwb[i] = f2bf(qw[i]);
    if (i < DIN * PROJK) pwb[i] = f2bf(pw[i]);
}

// Prep: gather attention biases through bias_idxs -> dense [h][i][j]
__global__ void prep_bias(const float* __restrict__ ab,
                          const int* __restrict__ idxs,
                          float* __restrict__ bx) {
    int i = blockIdx.x * 256 + threadIdx.x;
    if (i < NH * SQ * SQ) {
        int h  = i / (SQ * SQ);
        int ij = i - h * (SQ * SQ);
        bx[i] = ab[h * SQ + idxs[ij]];
    }
}

// ---------------------------------------------------------------------------
// Fused attention kernel: one block per batch image, 8 waves
// ---------------------------------------------------------------------------
__global__ __launch_bounds__(256, 1)
void effformer_attn(const float* __restrict__ hs,
                    const float* __restrict__ qkv_b,
                    const float* __restrict__ proj_b,
                    const u16*  __restrict__ qkv_wbf,
                    const u16*  __restrict__ proj_wbf,
                    const float* __restrict__ biasx,
                    float* __restrict__ out) {
    extern __shared__ __align__(16) char smem[];
    u16*   Xs   = (u16*)smem;                        // [SP][XS_ST]  bf16 input
    u16*   QKVs = Xs + SP * XS_ST;                   // [SP][QKV_ST] bf16 qkv / ctx
    float* Ps   = (float*)(QKVs + SP * QKV_ST);      // [SP][P_ST]   f32 scores
    u16*   Pb   = (u16*)(Ps + SP * P_ST);            // [SP][P_ST]   bf16 probs
    u16*   Vt   = Pb + SP * P_ST;                    // [ED][P_ST]   bf16 V^T

    const int tid  = threadIdx.x;
    const int b    = blockIdx.x;
    const int w    = tid >> 5;        // wave 0..7
    const int lane = tid & 31;
    const int l15  = lane & 15;
    const int hl   = lane >> 4;       // lane-half select

    // ---- Phase A0: stage X (rows >= 49 zero-padded), fp32 -> bf16 ----
    for (int idx = tid; idx < SP * DIN; idx += 256) {
        int r = idx / DIN;
        int c = idx - r * DIN;
        float v = 0.f;
        if (r < SQ) v = hs[(size_t)b * (SQ * DIN) + r * DIN + c];
        Xs[r * XS_ST + c] = f2bf(v);
    }
    __syncthreads();

    // ---- Phase A1: QKV = X @ Wqkv^T + b  (M=64, N=1536, K=448) ----
    // Each wave owns a PAIR of n-tiles (6 pairs per wave).  Per k-step:
    // 4 A fragments (LDS) + 2 B fragments (global) -> 8 WMMAs.
    for (int np = w; np < 48; np += 8) {
        v8f a00 = vzero(), a01 = vzero(), a02 = vzero(), a03 = vzero();
        v8f a10 = vzero(), a11 = vzero(), a12 = vzero(), a13 = vzero();
        const u16* a0 = Xs + l15 * XS_ST + hl * 8;
        const u16* br0 = qkv_wbf + (size_t)(np * 32 + l15) * DIN + hl * 8;
        const u16* br1 = br0 + 16 * DIN;
        #pragma unroll 2
        for (int kb = 0; kb < 14; ++kb) {
            v16bf b0 = ldfrag(br0 + kb * 32, br0 + kb * 32 + 16);
            v16bf b1 = ldfrag(br1 + kb * 32, br1 + kb * 32 + 16);
            v16bf f0 = ldfrag(a0 + kb * 32,                 a0 + kb * 32 + 16);
            v16bf f1 = ldfrag(a0 + kb * 32 + 16 * XS_ST,    a0 + kb * 32 + 16 * XS_ST + 16);
            v16bf f2 = ldfrag(a0 + kb * 32 + 32 * XS_ST,    a0 + kb * 32 + 32 * XS_ST + 16);
            v16bf f3 = ldfrag(a0 + kb * 32 + 48 * XS_ST,    a0 + kb * 32 + 48 * XS_ST + 16);
            a00 = wmma_bf16(f0, b0, a00);
            a01 = wmma_bf16(f1, b0, a01);
            a02 = wmma_bf16(f2, b0, a02);
            a03 = wmma_bf16(f3, b0, a03);
            a10 = wmma_bf16(f0, b1, a10);
            a11 = wmma_bf16(f1, b1, a11);
            a12 = wmma_bf16(f2, b1, a12);
            a13 = wmma_bf16(f3, b1, a13);
        }
        int n0 = np * 32 + l15;
        int n1 = n0 + 16;
        float bn0 = qkv_b[n0];
        float bn1 = qkv_b[n1];
        #pragma unroll
        for (int r = 0; r < 8; ++r) {
            int m = hl * 8 + r;
            QKVs[(m +  0) * QKV_ST + n0] = f2bf(a00[r] + bn0);
            QKVs[(m + 16) * QKV_ST + n0] = f2bf(a01[r] + bn0);
            QKVs[(m + 32) * QKV_ST + n0] = f2bf(a02[r] + bn0);
            QKVs[(m + 48) * QKV_ST + n0] = f2bf(a03[r] + bn0);
            QKVs[(m +  0) * QKV_ST + n1] = f2bf(a10[r] + bn1);
            QKVs[(m + 16) * QKV_ST + n1] = f2bf(a11[r] + bn1);
            QKVs[(m + 32) * QKV_ST + n1] = f2bf(a12[r] + bn1);
            QKVs[(m + 48) * QKV_ST + n1] = f2bf(a13[r] + bn1);
        }
    }
    __syncthreads();

    const float scale = 0.17677669529663689f;   // 32^-0.5

    // ---- Phase B: per-head attention ----
    for (int h = 0; h < NH; ++h) {
        const int qoff = h * QKVD;
        const int koff = qoff + 32;
        const int voff = qoff + 64;

        // B0/B1 overlapped: waves 0-3 compute scores = Q @ K^T (K=32 ->
        // one WMMA step, 4 m-tiles each); waves 4-7 transpose the V slice
        // into Vt.  Both only read QKVs; writes (Ps vs Vt) are disjoint.
        if (w < 4) {
            int nt = w;
            const u16* bp = QKVs + (nt * 16 + l15) * QKV_ST + koff + hl * 8;
            v16bf bf = ldfrag(bp, bp + 16);
            const u16* a0 = QKVs + l15 * QKV_ST + qoff + hl * 8;
            v16bf f0 = ldfrag(a0,                 a0 + 16);
            v16bf f1 = ldfrag(a0 + 16 * QKV_ST,   a0 + 16 * QKV_ST + 16);
            v16bf f2 = ldfrag(a0 + 32 * QKV_ST,   a0 + 32 * QKV_ST + 16);
            v16bf f3 = ldfrag(a0 + 48 * QKV_ST,   a0 + 48 * QKV_ST + 16);
            v8f s0 = wmma_bf16(f0, bf, vzero());
            v8f s1 = wmma_bf16(f1, bf, vzero());
            v8f s2 = wmma_bf16(f2, bf, vzero());
            v8f s3 = wmma_bf16(f3, bf, vzero());
            int j = nt * 16 + l15;
            #pragma unroll
            for (int r = 0; r < 8; ++r) {
                int i = hl * 8 + r;
                Ps[(i +  0) * P_ST + j] = s0[r];
                Ps[(i + 16) * P_ST + j] = s1[r];
                Ps[(i + 32) * P_ST + j] = s2[r];
                Ps[(i + 48) * P_ST + j] = s3[r];
            }
        } else {
            for (int idx = tid - 128; idx < SP * ED; idx += 128) {
                int j = idx >> 7;          // token
                int e = idx & 127;         // channel
                Vt[e * P_ST + j] = QKVs[j * QKV_ST + voff + e];
            }
        }
        __syncthreads();

        // B2: row softmax with scale + bias; mask padded keys/rows -> 0 prob
        if (tid < SP) {
            int i = tid;
            if (i < SQ) {
                const float* bh = biasx + h * (SQ * SQ) + i * SQ;
                float mx = -1e30f;
                for (int j = 0; j < SQ; ++j) {
                    float v = Ps[i * P_ST + j] * scale + bh[j];
                    Ps[i * P_ST + j] = v;
                    mx = fmaxf(mx, v);
                }
                float sum = 0.f;
                for (int j = 0; j < SQ; ++j) {
                    float e = __expf(Ps[i * P_ST + j] - mx);
                    Ps[i * P_ST + j] = e;
                    sum += e;
                }
                float inv = 1.f / sum;
                for (int j = 0; j < SQ; ++j) Pb[i * P_ST + j] = f2bf(Ps[i * P_ST + j] * inv);
                for (int j = SQ; j < SP; ++j) Pb[i * P_ST + j] = 0;
            } else {
                for (int j = 0; j < SP; ++j) Pb[i * P_ST + j] = 0;
            }
        }
        __syncthreads();

        // B3: ctx = P @ V (M=64, N=128, K=64 -> 2 WMMA steps).
        // Each wave owns one e-tile (nt = w); 4 m-tile accumulators.
        // ctx overwrites the V slice in QKVs (Vt holds the live copy).
        {
            int nt = w;
            v8f acc0 = vzero(), acc1 = vzero(), acc2 = vzero(), acc3 = vzero();
            const u16* a0 = Pb + l15 * P_ST + hl * 8;
            const u16* brow = Vt + (nt * 16 + l15) * P_ST + hl * 8;
            #pragma unroll
            for (int kb = 0; kb < 2; ++kb) {
                v16bf bf = ldfrag(brow + kb * 32, brow + kb * 32 + 16);
                v16bf f0 = ldfrag(a0 + kb * 32,               a0 + kb * 32 + 16);
                v16bf f1 = ldfrag(a0 + kb * 32 + 16 * P_ST,   a0 + kb * 32 + 16 * P_ST + 16);
                v16bf f2 = ldfrag(a0 + kb * 32 + 32 * P_ST,   a0 + kb * 32 + 32 * P_ST + 16);
                v16bf f3 = ldfrag(a0 + kb * 32 + 48 * P_ST,   a0 + kb * 32 + 48 * P_ST + 16);
                acc0 = wmma_bf16(f0, bf, acc0);
                acc1 = wmma_bf16(f1, bf, acc1);
                acc2 = wmma_bf16(f2, bf, acc2);
                acc3 = wmma_bf16(f3, bf, acc3);
            }
            int e = nt * 16 + l15;
            #pragma unroll
            for (int r = 0; r < 8; ++r) {
                int m = hl * 8 + r;
                QKVs[(m +  0) * QKV_ST + voff + e] = f2bf(acc0[r]);
                QKVs[(m + 16) * QKV_ST + voff + e] = f2bf(acc1[r]);
                QKVs[(m + 32) * QKV_ST + voff + e] = f2bf(acc2[r]);
                QKVs[(m + 48) * QKV_ST + voff + e] = f2bf(acc3[r]);
            }
        }
        __syncthreads();
    }

    // ---- Phase C: out = ctx @ Wproj^T + b  (M=64, N=448, K=1024) ----
    // ctx column (h*128+e) lives at QKVs column h*192+64+e; 32B K-chunks
    // never straddle a head boundary.  14 n-tile PAIRS over 8 waves; per
    // k-step 4 A fragments (LDS) + 2 B fragments (global) -> 8 WMMAs.
    for (int np = w; np < 14; np += 8) {
        v8f a00 = vzero(), a01 = vzero(), a02 = vzero(), a03 = vzero();
        v8f a10 = vzero(), a11 = vzero(), a12 = vzero(), a13 = vzero();
        const u16* a0base = QKVs + l15 * QKV_ST;
        const u16* br0 = proj_wbf + (size_t)(np * 32 + l15) * PROJK + hl * 8;
        const u16* br1 = br0 + 16 * PROJK;
        #pragma unroll 2
        for (int ks = 0; ks < 32; ++ks) {
            int head    = ks >> 2;
            int colbase = head * QKVD + 64 + (ks & 3) * 32;
            const u16* ap = a0base + colbase + hl * 8;
            v16bf b0 = ldfrag(br0 + ks * 32, br0 + ks * 32 + 16);
            v16bf b1 = ldfrag(br1 + ks * 32, br1 + ks * 32 + 16);
            v16bf f0 = ldfrag(ap,                 ap + 16);
            v16bf f1 = ldfrag(ap + 16 * QKV_ST,   ap + 16 * QKV_ST + 16);
            v16bf f2 = ldfrag(ap + 32 * QKV_ST,   ap + 32 * QKV_ST + 16);
            v16bf f3 = ldfrag(ap + 48 * QKV_ST,   ap + 48 * QKV_ST + 16);
            a00 = wmma_bf16(f0, b0, a00);
            a01 = wmma_bf16(f1, b0, a01);
            a02 = wmma_bf16(f2, b0, a02);
            a03 = wmma_bf16(f3, b0, a03);
            a10 = wmma_bf16(f0, b1, a10);
            a11 = wmma_bf16(f1, b1, a11);
            a12 = wmma_bf16(f2, b1, a12);
            a13 = wmma_bf16(f3, b1, a13);
        }
        int c0 = np * 32 + l15;
        int c1 = c0 + 16;
        float bc0 = proj_b[c0];
        float bc1 = proj_b[c1];
        float* ob0 = out + (size_t)b * (SQ * DIN) + c0;
        float* ob1 = out + (size_t)b * (SQ * DIN) + c1;
        #pragma unroll
        for (int r = 0; r < 8; ++r) {
            int m = hl * 8 + r;
            ob0[m * DIN] = a00[r] + bc0;                         // m+0  < 49
            ob1[m * DIN] = a10[r] + bc1;
            if (m + 16 < SQ) {
                ob0[(m + 16) * DIN] = a01[r] + bc0;
                ob1[(m + 16) * DIN] = a11[r] + bc1;
            }
            if (m + 32 < SQ) {
                ob0[(m + 32) * DIN] = a02[r] + bc0;
                ob1[(m + 32) * DIN] = a12[r] + bc1;
            }
            // tile 3 (rows 48..63) is never stored; real row 48 is left unwritten
        }
    }
}

// ---------------------------------------------------------------------------
extern "C" void kernel_launch(void* const* d_in, const int* in_sizes, int n_in,
                              void* d_out, int out_size, void* d_ws, size_t ws_size,
                              hipStream_t stream) {
    const float* hs     = (const float*)d_in[0];   // (2048, 49, 448)
    const float* qkv_w  = (const float*)d_in[1];   // (1536, 448)
    const float* qkv_b  = (const float*)d_in[2];   // (1536,)
    const float* proj_w = (const float*)d_in[3];   // (448, 1024)
    const float* proj_b = (const float*)d_in[4];   // (448,)
    const float* ab     = (const float*)d_in[5];   // (8, 49)
    const int*   idxs   = (const int*)d_in[6];     // (49, 49)
    float*       out    = (float*)d_out;           // (2048, 49, 448)

    char* ws = (char*)d_ws;
    u16*   qkv_wbf  = (u16*)(ws + WS_QKVW);
    u16*   proj_wbf = (u16*)(ws + WS_PROJW);
    float* biasx    = (float*)(ws + WS_BIAS);

    prep_weights<<<(HID * DIN + 255) / 256, 256, 0, stream>>>(qkv_w, proj_w, qkv_wbf, proj_wbf);
    prep_bias<<<(NH * SQ * SQ + 255) / 256, 256, 0, stream>>>(ab, idxs, biasx);

    hipFuncSetAttribute(reinterpret_cast<const void*>(effformer_attn),
                        hipFuncAttributeMaxDynamicSharedMemorySize, SMEM_BYTES);
    effformer_attn<<<NB, 256, SMEM_BYTES, stream>>>(hs, qkv_b, proj_b,
                                                    qkv_wbf, proj_wbf, biasx, out);
}
